// GraphAttentionEmbedding_146028888474
// MI455X (gfx1250) — matrix operand order repaired and checked by reading, not compiled, not run on hardware
//
#include <hip/hip_runtime.h>
#include <hip/hip_bf16.h>

// ---------------------------------------------------------------------------
// Temporal Graph Attention (TGAT, 2 layers) for MI455X / gfx1250.
// All dense contractions run through v_wmma_f32_16x16x32_f16 (f16 in, f32 acc).
// Weights are pre-transposed to [N][K] f16. GEMM uses 64x128 tiles, 8 waves
// (2x4), 4 WMMA per wave per K-step, LDS ping-pong double buffering with
// CDNA5 async memory->LDS copies (GLOBAL_LOAD_ASYNC_TO_LDS_B128, ASYNCcnt)
// and one barrier per K-step. Gathers are staged once into padded f16
// feature matrices; softmax/LN are small VALU epilogues. Work is chunked
// (1024 queries/chunk) so the workspace stays ~85 MB.
// ---------------------------------------------------------------------------

typedef _Float16 half_t;
typedef __attribute__((ext_vector_type(16))) _Float16 v16h;
typedef __attribute__((ext_vector_type(8)))  float    v8f;

#define BATCH   1024
#define KNBR    20
#define DN      172
#define DE      172
#define DT      100
#define QD      272      // DN + DT
#define KD      444      // DN + DE + DT
#define HD      136      // QD / NH
#define KD_PAD  448      // mult of 32
#define QD_PAD  288      // mult of 32
#define DN_PAD  176      // mult of 16
#define H1_PAD  192      // mult of 32 (fc1 hidden 172 -> 192)

static __device__ inline v8f wmma_f16(v16h a, v16h b, v8f c) {
  return __builtin_amdgcn_wmma_f32_16x16x32_f16(false, a, false, b,
                                                (short)0, c, false, false);
}

// CDNA5 async memory -> LDS copy (16B per lane), tracked by ASYNCcnt.
// VDST = LDS byte address (low 32 bits of the generic pointer), VADDR = u64.
static __device__ inline void async_ld_b128(unsigned lds_off, const void* gaddr) {
  asm volatile("global_load_async_to_lds_b128 %0, %1, off"
               :: "v"(lds_off), "v"(gaddr) : "memory");
}
static __device__ inline void wait_async0() {
  asm volatile("s_wait_asynccnt 0x0" ::: "memory");
}

// ===========================================================================
// Weight conversion f32 (K x N) -> padded, TRANSPOSED f16 (Np rows x Kp cols)
// ===========================================================================
__global__ void cvt_wt_kernel(const float* __restrict__ src, half_t* __restrict__ dst,
                              int Ks, int Ns, int Kp, int Np) {
  int k = blockIdx.x;                       // 0..Kp-1
  for (int n = threadIdx.x; n < Np; n += 128) {
    float v = (k < Ks && n < Ns) ? src[(size_t)k * Ns + n] : 0.f;
    dst[(size_t)n * Kp + k] = (half_t)v;    // transposed store
  }
}

// ===========================================================================
// Build KV feature rows: [conv(172) | edge(172) | time(100) | pad(4)] as f16
// ===========================================================================
__global__ __launch_bounds__(128) void build_kv_kernel(
    const int* __restrict__ nbr_ids, const int* __restrict__ nbr_eids,
    const float* __restrict__ q_times, const float* __restrict__ nbr_times,
    const float* __restrict__ memories, const float* __restrict__ node_raw,
    const float* __restrict__ edge_raw,
    const float* __restrict__ conv_rows,        // null -> conv0(id)
    const float* __restrict__ tw, const float* __restrict__ tb,
    half_t* __restrict__ Akv, int R) {
  int r = blockIdx.x;
  if (r >= R) return;
  int id  = nbr_ids[r];
  int eid = nbr_eids[r];
  float dt = q_times[r / KNBR] - nbr_times[r];
  half_t* dst = Akv + (size_t)r * KD_PAD;
  for (int c = threadIdx.x; c < KD_PAD; c += 128) {
    float v;
    if (c < DN) {
      v = conv_rows ? conv_rows[(size_t)r * DN + c]
                    : memories[(size_t)id * DN + c] + node_raw[(size_t)id * DN + c];
    } else if (c < DN + DE) {
      v = edge_raw[(size_t)eid * DE + (c - DN)];
    } else if (c < KD) {
      int j = c - DN - DE;
      v = __cosf(dt * tw[j] + tb[j]);
    } else {
      v = 0.f;
    }
    dst[c] = (half_t)v;
  }
}

// ===========================================================================
// Build query features: qA f16 (Mq x 288), q_in f32 (Mq x 272, residual),
// and the nf half of the merge input (mergeA cols 272..447).
// ===========================================================================
__global__ __launch_bounds__(128) void build_q_kernel(
    const int* __restrict__ q_ids,
    const float* __restrict__ q_conv_rows,      // null -> conv0(q_ids)
    const float* __restrict__ memories, const float* __restrict__ node_raw,
    const float* __restrict__ tw, const float* __restrict__ tb,
    half_t* __restrict__ qA, float* __restrict__ qin,
    half_t* __restrict__ mergeA, int Mq) {
  int m = blockIdx.x;
  if (m >= Mq) return;
  int id = q_ids[m];
  for (int c = threadIdx.x; c < QD_PAD; c += 128) {
    float v;
    if (c < DN)       v = q_conv_rows ? q_conv_rows[(size_t)m * DN + c]
                                      : memories[(size_t)id * DN + c] + node_raw[(size_t)id * DN + c];
    else if (c < QD)  v = __cosf(tb[c - DN]);           // time_encode(0)
    else              v = 0.f;
    qA[(size_t)m * QD_PAD + c] = (half_t)v;
    if (c < QD) qin[(size_t)m * QD + c] = v;
  }
  for (int c = threadIdx.x; c < DN_PAD; c += 128) {
    float v = (c < DN) ? memories[(size_t)id * DN + c] + node_raw[(size_t)id * DN + c] : 0.f;
    mergeA[(size_t)m * KD_PAD + QD + c] = (half_t)v;
  }
}

// ===========================================================================
// WMMA GEMM: C(f32) = A(f16, M x Kd) @ W (f16 transposed, nrows_b x Kd)
// BM=64, BN=128, BK=32; 256 threads = 8 waves in 2x4; each wave owns a
// 32x32 C block (2x2 WMMA tiles). Tiles stage via async global->LDS copies
// (ASYNCcnt) into ping-pong LDS buffers; one barrier per K-step.
// Fragment layouts follow CDNA5 ISA 7.12.2 (16-bit A/B, f32 C/D).
// ===========================================================================
__global__ __launch_bounds__(256) void gemm_f16_kernel(
    const half_t* __restrict__ A, int lda,
    const half_t* __restrict__ Bt, int ldbt, int nrows_b,
    int M, int N, int Kd,
    const float* __restrict__ bias,
    const float* __restrict__ resid, int ldres,
    float* __restrict__ Cf, int ldc,
    half_t* __restrict__ Ch, int ldch, int nzero,
    int relu) {
  __shared__ __align__(16) half_t As[2][64][40];    // 64 x 32 (+8 pad)
  __shared__ __align__(16) half_t Bs[2][128][40];   // [n][k] 128 x 32 (+8 pad)

  int tid  = threadIdx.x;
  int wave = tid >> 5;
  int lane = tid & 31;
  int m0 = blockIdx.y * 64;
  int n0 = blockIdx.x * 128;

  // staging coords: 16B per lane; A: 1 async b128, B: 2 async b128
  int arow = tid >> 2;              // 0..63
  int ak8  = (tid & 3) * 8;         // 0,8,16,24
  int nb0  = tid >> 2;              // B rows 0..63
  int nb1  = (tid + 256) >> 2;      // B rows 64..127
  // clamp out-of-range B rows (their products land in discarded columns)
  int nc0 = n0 + nb0; if (nc0 >= nrows_b) nc0 = nrows_b - 1;
  int nc1 = n0 + nb1; if (nc1 >= nrows_b) nc1 = nrows_b - 1;

  auto stage_async = [&](int b, int k0) {
    async_ld_b128((unsigned)(uintptr_t)&As[b][arow][ak8],
                  A + (size_t)(m0 + arow) * lda + (k0 + ak8));
    async_ld_b128((unsigned)(uintptr_t)&Bs[b][nb0][ak8],
                  Bt + (size_t)nc0 * ldbt + (k0 + ak8));
    async_ld_b128((unsigned)(uintptr_t)&Bs[b][nb1][ak8],
                  Bt + (size_t)nc1 * ldbt + (k0 + ak8));
  };

  v8f zacc = {};
  v8f acc[2][2];
#pragma unroll
  for (int i = 0; i < 2; ++i)
#pragma unroll
    for (int j = 0; j < 2; ++j) acc[i][j] = zacc;

  // prologue: stage tile 0
  stage_async(0, 0);
  wait_async0();
  __syncthreads();

  int wm = (wave >> 2) * 32;        // wave row base within block
  int wn = (wave & 3) * 32;         // wave col base within block
  int mlane = lane & 15;
  int koff  = (lane >> 4) * 8;      // A frag: lanes 0-15 K0..7/16..23; 16-31 K8..15/24..31
  int kb    = (lane >> 4) * 8;      // B frag: lanes 0-15 K0..7/16..23; 16-31 K8..15/24..31

  int buf = 0;
  for (int k0 = 0; k0 < Kd; k0 += 32) {
    bool nxt = (k0 + 32) < Kd;
    if (nxt) {
      stage_async(buf ^ 1, k0 + 32);            // async fill of next buffer
      if (k0 + 64 < Kd)
        __builtin_prefetch(A + (size_t)(m0 + arow) * lda + (k0 + 64 + ak8), 0, 3);
    }

    v16h a0, a1, b0, b1;
#pragma unroll
    for (int j = 0; j < 8; ++j) {
      a0[j]     = As[buf][wm + mlane][koff + j];
      a0[j + 8] = As[buf][wm + mlane][16 + koff + j];
      a1[j]     = As[buf][wm + 16 + mlane][koff + j];
      a1[j + 8] = As[buf][wm + 16 + mlane][16 + koff + j];
    }
#pragma unroll
    for (int j = 0; j < 8; ++j) {
      b0[j]     = Bs[buf][wn + mlane][kb + j];
      b0[j + 8] = Bs[buf][wn + mlane][16 + kb + j];
      b1[j]     = Bs[buf][wn + 16 + mlane][kb + j];
      b1[j + 8] = Bs[buf][wn + 16 + mlane][16 + kb + j];
    }

    acc[0][0] = wmma_f16(a0, b0, acc[0][0]);
    acc[0][1] = wmma_f16(a0, b1, acc[0][1]);
    acc[1][0] = wmma_f16(a1, b0, acc[1][0]);
    acc[1][1] = wmma_f16(a1, b1, acc[1][1]);

    if (nxt) {
      wait_async0();                            // async writes landed
      __syncthreads();                          // visible to all waves
      buf ^= 1;
    }
  }

  // epilogue: C layout VGPR r, lane L -> row (L<16 ? r : r+8), col L&15
#pragma unroll
  for (int mi = 0; mi < 2; ++mi) {
#pragma unroll
    for (int ni = 0; ni < 2; ++ni) {
      int colg = n0 + wn + ni * 16 + mlane;
#pragma unroll
      for (int r = 0; r < 8; ++r) {
        int rowg = m0 + wm + mi * 16 + ((lane < 16) ? r : (r + 8));
        float v = acc[mi][ni][r];
        if (colg < N) {
          if (bias)  v += bias[colg];
          if (resid) v += resid[(size_t)rowg * ldres + colg];
          if (relu)  v = fmaxf(v, 0.f);
          if (Cf && rowg < M) Cf[(size_t)rowg * ldc + colg] = v;
          if (Ch && rowg < M) Ch[(size_t)rowg * ldch + colg] = (half_t)v;
        } else if (Ch && colg < nzero && rowg < M) {
          Ch[(size_t)rowg * ldch + colg] = (half_t)0.f;   // zero f16 padding
        }
      }
    }
  }
}

// ===========================================================================
// Masked softmax attention: one wave per (query, head). K=20 neighbors.
// ===========================================================================
__global__ __launch_bounds__(128) void attn_kernel(
    const float* __restrict__ Qm, const float* __restrict__ Km,
    const float* __restrict__ Vm, const int* __restrict__ nbr_ids,
    half_t* __restrict__ attA, int Mq) {
  __shared__ float satt[4][KNBR];
  int w    = threadIdx.x >> 5;
  int lane = threadIdx.x & 31;
  int gw   = blockIdx.x * 4 + w;
  int m = gw >> 1;
  int h = gw & 1;
  if (m >= Mq) return;

  const float* q = Qm + (size_t)m * QD + h * HD;
  float score = -1e30f;
  if (lane < KNBR) {
    int r = m * KNBR + lane;
    const float* kp = Km + (size_t)r * QD + h * HD;
    float s = 0.f;
    for (int d = 0; d < HD; ++d) s += q[d] * kp[d];
    s *= 0.08574929257125442f;                 // HD^-0.5
    if (nbr_ids[r] == 0) s = -1e10f;           // padded neighbor mask
    score = s;
  }
  float mx = score;
  for (int o = 16; o > 0; o >>= 1) mx = fmaxf(mx, __shfl_xor(mx, o, 32));
  float e = (lane < KNBR) ? __expf(score - mx) : 0.f;
  float se = e;
  for (int o = 16; o > 0; o >>= 1) se += __shfl_xor(se, o, 32);
  float att = e / se;
  if (lane < KNBR) satt[w][lane] = att;
  __syncthreads();

  for (int d = lane; d < HD; d += 32) {
    float acc = 0.f;
    const float* vp = Vm + (size_t)(m * KNBR) * QD + h * HD + d;
    for (int k = 0; k < KNBR; ++k) acc += satt[w][k] * vp[(size_t)k * QD];
    attA[(size_t)m * QD_PAD + h * HD + d] = (half_t)acc;
  }
  if (h == 0 && lane < QD_PAD - QD)
    attA[(size_t)m * QD_PAD + QD + lane] = (half_t)0.f;
}

// ===========================================================================
// LayerNorm over 272 cols; writes f16 into mergeA cols 0..271. 1 wave / row.
// ===========================================================================
__global__ __launch_bounds__(256) void ln_kernel(
    const float* __restrict__ x, const float* __restrict__ g,
    const float* __restrict__ be, half_t* __restrict__ mergeA, int Mq) {
  int row  = blockIdx.x * 8 + (threadIdx.x >> 5);
  int lane = threadIdx.x & 31;
  if (row >= Mq) return;
  const float* xr = x + (size_t)row * QD;
  float s = 0.f, s2 = 0.f;
  for (int c = lane; c < QD; c += 32) { float v = xr[c]; s += v; s2 += v * v; }
  for (int o = 16; o > 0; o >>= 1) { s += __shfl_xor(s, o, 32); s2 += __shfl_xor(s2, o, 32); }
  float mu   = s  * (1.f / QD);
  float var  = s2 * (1.f / QD) - mu * mu;
  float rstd = rsqrtf(var + 1e-5f);
  for (int c = lane; c < QD; c += 32) {
    float v = (xr[c] - mu) * rstd * g[c] + be[c];
    mergeA[(size_t)row * KD_PAD + c] = (half_t)v;
  }
}

// ===========================================================================
// Host-side orchestration
// ===========================================================================
static inline void launch_gemm(hipStream_t s, const half_t* A, int lda,
                               const half_t* Bt, int ldbt, int nrows_b,
                               int M, int N, int Kd,
                               const float* bias, const float* resid, int ldres,
                               float* Cf, int ldc, half_t* Ch, int ldch,
                               int nzero, int relu) {
  dim3 grid((N + 127) / 128, M / 64);
  gemm_f16_kernel<<<grid, 256, 0, s>>>(A, lda, Bt, ldbt, nrows_b, M, N, Kd,
                                       bias, resid, ldres, Cf, ldc, Ch, ldch,
                                       nzero, relu);
}

struct Weights16 {           // all weight matrices stored transposed [N][Kpad]
  const half_t *Wq, *Wk, *Wv, *Wo, *W1, *W2;
  const float  *bo, *g, *be, *b1, *b2;
};

struct Scratch {
  half_t* Akv;   float* Kmat;  float* Vmat;
  half_t* qA;    float* qin;   float* Qmat;
  half_t* attA;  float* xbuf;  half_t* mergeA; half_t* hbuf;
};

// One temporal-conv layer over Mq queries (Mq multiple of 64)
static void run_layer(hipStream_t s, int Mq,
                      const int* q_ids, const int* nbr_ids, const int* nbr_eids,
                      const float* q_times, const float* nbr_times,
                      const float* kv_conv_rows, const float* q_conv_rows,
                      const float* memories, const float* node_raw,
                      const float* edge_raw, const float* tw, const float* tb,
                      const Weights16& W, const Scratch& S, float* outc) {
  int R = Mq * KNBR;
  build_kv_kernel<<<R, 128, 0, s>>>(nbr_ids, nbr_eids, q_times, nbr_times,
                                    memories, node_raw, edge_raw, kv_conv_rows,
                                    tw, tb, S.Akv, R);
  build_q_kernel<<<Mq, 128, 0, s>>>(q_ids, q_conv_rows, memories, node_raw,
                                    tw, tb, S.qA, S.qin, S.mergeA, Mq);
  // K = kv @ Wk ; V = kv @ Wv
  launch_gemm(s, S.Akv, KD_PAD, W.Wk, KD_PAD, QD, R, QD, KD_PAD,
              nullptr, nullptr, 0, S.Kmat, QD, nullptr, 0, 0, 0);
  launch_gemm(s, S.Akv, KD_PAD, W.Wv, KD_PAD, QD, R, QD, KD_PAD,
              nullptr, nullptr, 0, S.Vmat, QD, nullptr, 0, 0, 0);
  // Q = q_in @ Wq
  launch_gemm(s, S.qA, QD_PAD, W.Wq, QD_PAD, QD, Mq, QD, QD_PAD,
              nullptr, nullptr, 0, S.Qmat, QD, nullptr, 0, 0, 0);
  // masked softmax attention
  attn_kernel<<<(Mq * 2) / 4, 128, 0, s>>>(S.Qmat, S.Kmat, S.Vmat, nbr_ids,
                                           S.attA, Mq);
  // x = att_out @ Wo + bo + q_in
  launch_gemm(s, S.attA, QD_PAD, W.Wo, QD_PAD, QD, Mq, QD, QD_PAD,
              W.bo, S.qin, QD, S.xbuf, QD, nullptr, 0, 0, 0);
  // LayerNorm -> mergeA[:, 0:272]
  ln_kernel<<<Mq / 8, 256, 0, s>>>(S.xbuf, W.g, W.be, S.mergeA, Mq);
  // h = relu(mergeA @ w1 + b1) (f16 out, padded to 192)
  launch_gemm(s, S.mergeA, KD_PAD, W.W1, KD_PAD, DN_PAD, Mq, DN, KD_PAD,
              W.b1, nullptr, 0, nullptr, 0, S.hbuf, H1_PAD, H1_PAD, 1);
  // out = h @ w2 + b2
  launch_gemm(s, S.hbuf, H1_PAD, W.W2, H1_PAD, DN_PAD, Mq, DN, H1_PAD,
              W.b2, nullptr, 0, outc, DN, nullptr, 0, 0, 0);
}

extern "C" void kernel_launch(void* const* d_in, const int* in_sizes, int n_in,
                              void* d_out, int out_size, void* d_ws, size_t ws_size,
                              hipStream_t stream) {
  const float* node_raw = (const float*)d_in[0];
  const float* edge_raw = (const float*)d_in[1];
  const float* memories = (const float*)d_in[2];
  const float* t_node   = (const float*)d_in[3];
  const float* t_n1     = (const float*)d_in[4];
  const float* t_n2     = (const float*)d_in[5];
  const float* time_w   = (const float*)d_in[6];
  const float* time_b   = (const float*)d_in[7];
  const float* Wq       = (const float*)d_in[8];
  const float* Wk       = (const float*)d_in[9];
  const float* Wv       = (const float*)d_in[10];
  const float* Wo       = (const float*)d_in[11];
  const float* bo       = (const float*)d_in[12];
  const float* ln_g     = (const float*)d_in[13];
  const float* ln_b     = (const float*)d_in[14];
  const float* m_w1     = (const float*)d_in[15];
  const float* m_b1     = (const float*)d_in[16];
  const float* m_w2     = (const float*)d_in[17];
  const float* m_b2     = (const float*)d_in[18];
  const int* node_ids   = (const int*)d_in[19];
  const int* n1_ids     = (const int*)d_in[20];
  const int* n1_eids    = (const int*)d_in[21];
  const int* n2_ids     = (const int*)d_in[22];
  const int* n2_eids    = (const int*)d_in[23];

  // ---- workspace layout (256B aligned)
  char* p = (char*)d_ws;
  auto alloc = [&](size_t bytes) -> char* {
    char* q = p;
    p += (bytes + 255) & ~(size_t)255;
    return q;
  };
  half_t* Wq16[2], *Wk16[2], *Wv16[2], *Wo16[2], *W116[2], *W216[2];
  for (int l = 0; l < 2; ++l) {
    Wq16[l] = (half_t*)alloc((size_t)QD * QD_PAD * 2);      // [272][288]
    Wk16[l] = (half_t*)alloc((size_t)QD * KD_PAD * 2);      // [272][448]
    Wv16[l] = (half_t*)alloc((size_t)QD * KD_PAD * 2);
    Wo16[l] = (half_t*)alloc((size_t)QD * QD_PAD * 2);
    W116[l] = (half_t*)alloc((size_t)DN_PAD * KD_PAD * 2);  // [176][448]
    W216[l] = (half_t*)alloc((size_t)DN_PAD * H1_PAD * 2);  // [176][192]
  }
  float* nbr1_conv  = (float*)alloc((size_t)BATCH * KNBR * DN * 4);
  float* node1_conv = (float*)alloc((size_t)BATCH * DN * 4);

  const int MQ = 1024;              // queries per chunk
  const int RC = MQ * KNBR;         // 20480 kv rows per chunk
  Scratch S;
  S.Akv    = (half_t*)alloc((size_t)RC * KD_PAD * 2);
  S.Kmat   = (float*) alloc((size_t)RC * QD * 4);
  S.Vmat   = (float*) alloc((size_t)RC * QD * 4);
  S.qA     = (half_t*)alloc((size_t)MQ * QD_PAD * 2);
  S.qin    = (float*) alloc((size_t)MQ * QD * 4);
  S.Qmat   = (float*) alloc((size_t)MQ * QD * 4);
  S.attA   = (half_t*)alloc((size_t)MQ * QD_PAD * 2);
  S.xbuf   = (float*) alloc((size_t)MQ * QD * 4);
  S.mergeA = (half_t*)alloc((size_t)MQ * KD_PAD * 2);
  S.hbuf   = (half_t*)alloc((size_t)MQ * H1_PAD * 2);

  // ---- one-time weight conversion to padded transposed f16
  for (int l = 0; l < 2; ++l) {
    cvt_wt_kernel<<<QD_PAD, 128, 0, stream>>>(Wq + (size_t)l * QD * QD, Wq16[l], QD, QD, QD_PAD, QD);
    cvt_wt_kernel<<<KD_PAD, 128, 0, stream>>>(Wk + (size_t)l * KD * QD, Wk16[l], KD, QD, KD_PAD, QD);
    cvt_wt_kernel<<<KD_PAD, 128, 0, stream>>>(Wv + (size_t)l * KD * QD, Wv16[l], KD, QD, KD_PAD, QD);
    cvt_wt_kernel<<<QD_PAD, 128, 0, stream>>>(Wo + (size_t)l * QD * QD, Wo16[l], QD, QD, QD_PAD, QD);
    cvt_wt_kernel<<<KD_PAD, 128, 0, stream>>>(m_w1 + (size_t)l * KD * DN, W116[l], KD, DN, KD_PAD, DN_PAD);
    cvt_wt_kernel<<<H1_PAD, 128, 0, stream>>>(m_w2 + (size_t)l * DN * DN, W216[l], DN, DN, H1_PAD, DN_PAD);
  }

  Weights16 L0 = { Wq16[0], Wk16[0], Wv16[0], Wo16[0], W116[0], W216[0],
                   bo, ln_g, ln_b, m_b1, m_b2 };
  Weights16 L1 = { Wq16[1], Wk16[1], Wv16[1], Wo16[1], W116[1], W216[1],
                   bo + QD, ln_g + QD, ln_b + QD, m_b1 + DN, m_b2 + DN };

  // ---- Pass 1: layer-1 over all B*K flattened neighbors (20 chunks)
  const int NCHUNK = (BATCH * KNBR) / MQ;       // 20
  for (int c = 0; c < NCHUNK; ++c) {
    run_layer(stream, MQ,
              n1_ids + (size_t)c * MQ,
              n2_ids + (size_t)c * MQ * KNBR,
              n2_eids + (size_t)c * MQ * KNBR,
              t_n1 + (size_t)c * MQ,
              t_n2 + (size_t)c * MQ * KNBR,
              nullptr, nullptr,
              memories, node_raw, edge_raw, time_w, time_b,
              L0, S, nbr1_conv + (size_t)c * MQ * DN);
  }

  // ---- Pass 2: layer-1 over the B root nodes
  run_layer(stream, BATCH, node_ids, n1_ids, n1_eids, t_node, t_n1,
            nullptr, nullptr, memories, node_raw, edge_raw, time_w, time_b,
            L0, S, node1_conv);

  // ---- Pass 3: layer-2 (conv features come from layer-1 outputs)
  run_layer(stream, BATCH, node_ids, n1_ids, n1_eids, t_node, t_n1,
            nbr1_conv, node1_conv,
            memories, node_raw, edge_raw, time_w, time_b,
            L1, S, (float*)d_out);

  (void)in_sizes; (void)n_in; (void)out_size; (void)ws_size;
}
